// RNNModel_84834194031372
// MI455X (gfx1250) — matrix operand order repaired and checked by reading, not compiled, not run on hardware
//
#include <hip/hip_runtime.h>
#include <hip/hip_bf16.h>
#include <math.h>

typedef __attribute__((ext_vector_type(16))) __bf16 v16bf;
typedef __attribute__((ext_vector_type(8)))  __bf16 v8bf;
typedef __attribute__((ext_vector_type(8)))  float  v8f;

#define B_   1024
#define T_   100
#define D_   300
#define DP_  320      // D padded to multiple of 32 for K-chunks
#define U_   512
#define N4U  2048     // 4*U
#define C_   14
#define CHUNK_ELEMS (128*32*16)   // one K-chunk of swizzled weights (bf16 elems)

__device__ __forceinline__ float sigf(float x){ return 1.0f/(1.0f + __expf(-x)); }

// ------------------------------ prep kernels ------------------------------

__global__ void k_zero_u32(unsigned int* __restrict__ p, int n){
  int i = blockIdx.x*256 + threadIdx.x;
  if (i < n) p[i] = 0u;
}

__global__ void k_convert_x(const float* __restrict__ x, __bf16* __restrict__ xp, int total){
  int i = blockIdx.x*256 + threadIdx.x;
  if (i >= total) return;
  int k  = i % DP_;
  int bt = i / DP_;
  float v = (k < D_) ? x[(size_t)bt*D_ + k] : 0.0f;
  xp[i] = (__bf16)v;
}

// Swizzle a [K, 2048] f32 weight (optionally two stacked sources) into WMMA
// B-fragment tiles: layout [kc][nt][lane][16] bf16, where element i of lane L
// holds (k = kc*32 + (i<8 ? 8*(L>>4)+i : 16+8*(L>>4)+i-8), n = nt*16 + (L&15)).
__global__ void k_swizzle_w(const float* __restrict__ s0, const float* __restrict__ s1,
                            int ksplit, int kvalid, int total, __bf16* __restrict__ dst){
  int idx = blockIdx.x*256 + threadIdx.x;
  if (idx >= total) return;
  int i    = idx & 15;
  int lane = (idx >> 4) & 31;
  int tl   = idx >> 9;
  int nt   = tl & 127;
  int kc   = tl >> 7;
  int k = kc*32 + ((i < 8) ? (8*(lane >> 4) + i) : (16 + 8*(lane >> 4) + (i - 8)));
  int n = nt*16 + (lane & 15);
  float v = 0.0f;
  if (k < kvalid)
    v = (k < ksplit) ? s0[(size_t)k*N4U + n] : s1[(size_t)(k - ksplit)*N4U + n];
  dst[idx] = (__bf16)v;
}

// ------------------------- time-parallel input GEMM ------------------------
// Z0pre[t,b,:] = x[b,t,:] @ W0   (M = T*B rows, N = 2048, K = 320 padded)
// Output stored bf16 in raw WMMA C-fragment order: [t*64+btile][nt][lane][8].
__global__ __launch_bounds__(256, 1)
void k_gemm0(const __bf16* __restrict__ xp, const __bf16* __restrict__ wsw,
             __bf16* __restrict__ zpre){
  const int lane  = threadIdx.x & 31;
  const int w     = threadIdx.x >> 5;
  const int half  = lane >> 4;
  const int ln    = lane & 15;
  const int blk   = blockIdx.x;
  const int t     = blk >> 6;    // blk / 64
  const int btile = blk & 63;

  v8f acc[16];
#pragma unroll
  for (int m=0;m<16;++m)
#pragma unroll
    for (int e=0;e<8;++e) acc[m][e] = 0.0f;

  const __bf16* arow = xp + (size_t)((btile*16 + ln)*T_ + t) * DP_;

  // software-pipelined B-fragment double buffer (two 8-tile half-groups)
  v16bf b0[8], b1[8];
#pragma unroll
  for (int m=0;m<8;++m){
    const int nt = w*16 + m;
    b0[m] = *(const v16bf*)(wsw + ((size_t)nt*32 + lane)*16);
  }

  for (int kc = 0; kc < DP_/32; ++kc){
    const __bf16* wk  = wsw + (size_t)kc*CHUNK_ELEMS;
    const __bf16* wkn = wk + CHUNK_ELEMS;
    __builtin_prefetch(wkn + CHUNK_ELEMS, 0, 0);

    const int kb = kc*32 + half*8;
    v8bf alo = *(const v8bf*)(arow + kb);
    v8bf ahi = *(const v8bf*)(arow + kb + 16);
    v16bf a;
#pragma unroll
    for (int e=0;e<8;++e){ a[e] = alo[e]; a[e+8] = ahi[e]; }

    // issue loads for second half-group of this chunk
#pragma unroll
    for (int m=0;m<8;++m){
      const int nt = w*16 + 8 + m;
      b1[m] = *(const v16bf*)(wk + ((size_t)nt*32 + lane)*16);
    }
    // consume first half-group
#pragma unroll
    for (int m=0;m<8;++m)
      acc[m] = __builtin_amdgcn_wmma_f32_16x16x32_bf16(
          false, a, false, b0[m], (short)0, acc[m], false, false);
    // issue loads for first half-group of next chunk
    if (kc + 1 < DP_/32){
#pragma unroll
      for (int m=0;m<8;++m){
        const int nt = w*16 + m;
        b0[m] = *(const v16bf*)(wkn + ((size_t)nt*32 + lane)*16);
      }
    }
    // consume second half-group
#pragma unroll
    for (int m=0;m<8;++m)
      acc[8+m] = __builtin_amdgcn_wmma_f32_16x16x32_bf16(
          false, a, false, b1[m], (short)0, acc[8+m], false, false);
  }

  __bf16* zt = zpre + (size_t)(t*64 + btile)*(128*32*8);
#pragma unroll
  for (int m=0;m<16;++m){
    const int nt = w*16 + m;
    v8bf o;
#pragma unroll
    for (int e=0;e<8;++e) o[e] = (__bf16)acc[m][e];
    *(v8bf*)(zt + ((size_t)nt*32 + lane)*8) = o;
  }
}

// ------------------------- fused recurrent step ----------------------------
// One WG = 16 batch rows x full 2048 gate columns. 8 waves; each wave owns,
// for every gate g, 4 column-tiles: nt = g*32 + w*4 + j  (u-slice of 64).
// All four gates for a given (row,u) are lane-local after the GEMM.
// LayerNorm per (row,gate) reduces across waves through LDS.
// acc tile index m = g*4 + j.
__global__ __launch_bounds__(256, 1)
void k_step(const __bf16* __restrict__ xin, int xstride, int KC,
            const __bf16* __restrict__ wsw,
            const __bf16* __restrict__ zpre,              // per-t fragment base, or nullptr
            const float* __restrict__ gamma, const float* __restrict__ beta,
            const float* __restrict__ bias,
            const int* __restrict__ mask, int t,
            float* __restrict__ hstate, float* __restrict__ cstate,
            __bf16* __restrict__ hbf){                    // row stride 1024
  const int lane  = threadIdx.x & 31;
  const int w     = threadIdx.x >> 5;
  const int half  = lane >> 4;
  const int ln    = lane & 15;
  const int btile = blockIdx.x;

  v8f acc[16];
  if (zpre != nullptr){
    const __bf16* zt = zpre + (size_t)btile*(128*32*8);
#pragma unroll
    for (int m=0; m<16; ++m){
      const int nt = (m>>2)*32 + w*4 + (m&3);
      v8bf z = *(const v8bf*)(zt + ((size_t)nt*32 + lane)*8);
#pragma unroll
      for (int e=0;e<8;++e) acc[m][e] = (float)z[e];
    }
  } else {
#pragma unroll
    for (int m=0; m<16; ++m)
#pragma unroll
      for (int e=0;e<8;++e) acc[m][e] = 0.0f;
  }

  // GEMM: acc += Xin(16 x 32k) @ W(32k x 2048), double-buffered B fragments
  const __bf16* arow = xin + (size_t)(btile*16 + ln) * xstride;

  v16bf b0[8], b1[8];
#pragma unroll
  for (int m=0;m<8;++m){
    const int nt = (m>>2)*32 + w*4 + (m&3);        // tiles m = 0..7 (gates 0,1)
    b0[m] = *(const v16bf*)(wsw + ((size_t)nt*32 + lane)*16);
  }

  for (int kc = 0; kc < KC; ++kc){
    const __bf16* wk  = wsw + (size_t)kc*CHUNK_ELEMS;
    const __bf16* wkn = wk + CHUNK_ELEMS;
    __builtin_prefetch(wkn + CHUNK_ELEMS, 0, 0);

    const int kb = kc*32 + half*8;
    v8bf alo = *(const v8bf*)(arow + kb);
    v8bf ahi = *(const v8bf*)(arow + kb + 16);
    v16bf a;
#pragma unroll
    for (int e=0;e<8;++e){ a[e] = alo[e]; a[e+8] = ahi[e]; }

    // issue loads for tiles 8..15 (gates 2,3) of this chunk
#pragma unroll
    for (int m=0;m<8;++m){
      const int mm = m + 8;
      const int nt = (mm>>2)*32 + w*4 + (mm&3);
      b1[m] = *(const v16bf*)(wk + ((size_t)nt*32 + lane)*16);
    }
    // consume tiles 0..7
#pragma unroll
    for (int m=0;m<8;++m)
      acc[m] = __builtin_amdgcn_wmma_f32_16x16x32_bf16(
          false, a, false, b0[m], (short)0, acc[m], false, false);
    // issue loads for tiles 0..7 of the next chunk
    if (kc + 1 < KC){
#pragma unroll
      for (int m=0;m<8;++m){
        const int nt = (m>>2)*32 + w*4 + (m&3);
        b0[m] = *(const v16bf*)(wkn + ((size_t)nt*32 + lane)*16);
      }
    }
    // consume tiles 8..15
#pragma unroll
    for (int m=0;m<8;++m)
      acc[8+m] = __builtin_amdgcn_wmma_f32_16x16x32_bf16(
          false, a, false, b1[m], (short)0, acc[8+m], false, false);
  }

  // ---- LayerNorm per (row, gate) over 512 columns ----
  __shared__ float red[8][4][16][2];
#pragma unroll
  for (int g=0; g<4; ++g){
    float s[8], q[8];
#pragma unroll
    for (int v=0; v<8; ++v){ s[v]=0.0f; q[v]=0.0f; }
#pragma unroll
    for (int j=0; j<4; ++j)
#pragma unroll
      for (int v=0; v<8; ++v){
        float xv = acc[g*4+j][v];
        s[v] += xv; q[v] += xv*xv;
      }
    // reduce over the 16 columns held by this lane's half-wave
    for (int m=1; m<16; m<<=1){
#pragma unroll
      for (int v=0; v<8; ++v){
        s[v] += __shfl_xor(s[v], m, 32);
        q[v] += __shfl_xor(q[v], m, 32);
      }
    }
    if (ln == 0){
#pragma unroll
      for (int v=0; v<8; ++v){
        red[w][g][half*8+v][0] = s[v];
        red[w][g][half*8+v][1] = q[v];
      }
    }
  }
  __syncthreads();

#pragma unroll
  for (int g=0; g<4; ++g){
    float mu[8], rs[8];
#pragma unroll
    for (int v=0; v<8; ++v){
      const int row = half*8 + v;
      float s = 0.0f, q = 0.0f;
      for (int ww=0; ww<8; ++ww){ s += red[ww][g][row][0]; q += red[ww][g][row][1]; }
      float m   = s * (1.0f/512.0f);
      float var = q * (1.0f/512.0f) - m*m;
      mu[v] = m;
      rs[v] = rsqrtf(var + 1.0e-3f);
    }
#pragma unroll
    for (int j=0; j<4; ++j){
      const int u = w*64 + j*16 + ln;
      const float ga = gamma[g*U_ + u];
      const float be = beta[g*U_ + u];
      const float bi = bias[g*U_ + u];
#pragma unroll
      for (int v=0; v<8; ++v)
        acc[g*4+j][v] = ga*(acc[g*4+j][v] - mu[v])*rs[v] + be + bi;
    }
  }

  // ---- gates + masked state update ----
#pragma unroll
  for (int v=0; v<8; ++v){
    const int row = btile*16 + half*8 + v;
    const int mk  = mask[(size_t)row*T_ + t];
#pragma unroll
    for (int j=0; j<4; ++j){
      const int u = w*64 + j*16 + ln;
      float ig = sigf(acc[0*4+j][v]);
      float fg = sigf(acc[1*4+j][v]);
      float cg = tanhf(acc[2*4+j][v]);
      float og = sigf(acc[3*4+j][v]);
      const size_t si = (size_t)row*U_ + u;
      float cold = cstate[si];
      float hold = hstate[si];
      float cn = fg*cold + ig*cg;
      float hn = og*tanhf(cn);
      if (mk == 0){ cn = cold; hn = hold; }
      cstate[si] = cn;
      hstate[si] = hn;
      hbf[(size_t)row*1024 + u] = (__bf16)hn;
    }
  }
}

// ------------------------------ softmax head -------------------------------
__global__ __launch_bounds__(256)
void k_head(const float* __restrict__ h1, const float* __restrict__ Wd,
            const float* __restrict__ bd, float* __restrict__ out){
  const int lane = threadIdx.x & 31;
  const int w    = threadIdx.x >> 5;
  const int row  = blockIdx.x*8 + w;
  const bool valid = (lane < C_);
  float acc = valid ? bd[lane] : 0.0f;
  const float* hr = h1 + (size_t)row*U_;
  for (int k=0; k<U_; ++k){
    float hv = hr[k];
    if (valid) acc += hv * Wd[(size_t)k*C_ + lane];
  }
  float lg = valid ? acc : -3.0e38f;
  float mx = lg;
  for (int m=1; m<32; m<<=1) mx = fmaxf(mx, __shfl_xor(mx, m, 32));
  float e  = valid ? __expf(acc - mx) : 0.0f;
  float sm = e;
  for (int m=1; m<32; m<<=1) sm += __shfl_xor(sm, m, 32);
  if (valid) out[(size_t)row*C_ + lane] = e / sm;
}

// ------------------------------ host launcher ------------------------------
extern "C" void kernel_launch(void* const* d_in, const int* in_sizes, int n_in,
                              void* d_out, int out_size, void* d_ws, size_t ws_size,
                              hipStream_t stream){
  const float* x   = (const float*)d_in[0];
  const int*   msk = (const int*)  d_in[1];
  const float* W0  = (const float*)d_in[2];
  const float* U0  = (const float*)d_in[3];
  const float* b0  = (const float*)d_in[4];
  const float* g0  = (const float*)d_in[5];
  const float* be0 = (const float*)d_in[6];
  const float* W1  = (const float*)d_in[7];
  const float* U1  = (const float*)d_in[8];
  const float* b1  = (const float*)d_in[9];
  const float* g1  = (const float*)d_in[10];
  const float* be1 = (const float*)d_in[11];
  const float* Wd  = (const float*)d_in[12];
  const float* bd  = (const float*)d_in[13];
  float* out = (float*)d_out;

  char* ws = (char*)d_ws;
  size_t off = 0;
  auto take = [&](size_t bytes)->char*{
    char* p = ws + off;
    off = (off + bytes + 255) & ~(size_t)255;
    return p;
  };
  __bf16* xpad   = (__bf16*)take((size_t)B_*T_*DP_*2);      // 65.5 MB
  __bf16* w0sw   = (__bf16*)take((size_t)DP_*N4U*2);        // 1.3 MB
  __bf16* u0sw   = (__bf16*)take((size_t)U_*N4U*2);         // 2.1 MB
  __bf16* w1u1sw = (__bf16*)take((size_t)1024*N4U*2);       // 4.2 MB
  __bf16* zpre   = (__bf16*)take((size_t)T_*B_*N4U*2);      // 419 MB
  float*  h0     = (float*) take((size_t)B_*U_*4);
  float*  c0     = (float*) take((size_t)B_*U_*4);
  float*  h1     = (float*) take((size_t)B_*U_*4);
  float*  c1     = (float*) take((size_t)B_*U_*4);
  __bf16* h01    = (__bf16*)take((size_t)B_*1024*2);        // [h0_bf16 | h1_bf16]
  (void)ws_size; (void)in_sizes; (void)n_in; (void)out_size;

  // zero the (contiguous) state region: h0,c0,h1,c1 (f32) + h01 (bf16)
  {
    int words = (4*B_*U_*4 + B_*1024*2) / 4;
    k_zero_u32<<<(words+255)/256, 256, 0, stream>>>((unsigned int*)h0, words);
  }
  { int tot = B_*T_*DP_;
    k_convert_x<<<(tot+255)/256, 256, 0, stream>>>(x, xpad, tot); }
  { int tot = DP_*N4U;   // W0 padded to K=320 with zeros
    k_swizzle_w<<<(tot+255)/256, 256, 0, stream>>>(W0, W0, DP_, D_, tot, w0sw); }
  { int tot = U_*N4U;
    k_swizzle_w<<<(tot+255)/256, 256, 0, stream>>>(U0, U0, U_, U_, tot, u0sw); }
  { int tot = 1024*N4U;  // [W1 ; U1] stacked: K=1024
    k_swizzle_w<<<(tot+255)/256, 256, 0, stream>>>(W1, U1, U_, 1024, tot, w1u1sw); }

  // time-parallel input projection
  k_gemm0<<<T_*64, 256, 0, stream>>>(xpad, w0sw, zpre);

  // sequential recurrence
  for (int t = 0; t < T_; ++t){
    // layer 0: z = Z0pre[t] + h0 @ U0   (reads h01[:, :512], writes it back)
    k_step<<<64, 256, 0, stream>>>(h01, 1024, U_/32, u0sw,
                                   zpre + (size_t)t*B_*N4U,
                                   g0, be0, b0, msk, t, h0, c0, h01);
    // layer 1: z = [h0_new ; h1] @ [W1 ; U1]
    k_step<<<64, 256, 0, stream>>>(h01, 1024, 1024/32, w1u1sw,
                                   (const __bf16*)nullptr,
                                   g1, be1, b1, msk, t, h1, c1, h01 + U_);
  }

  // dense softmax head on final h1
  k_head<<<B_/8, 256, 0, stream>>>(h1, Wd, bd, out);
}
